// LSHSampledLayer_30588757082166
// MI455X (gfx1250) — matrix operand order repaired and checked by reading, not compiled, not run on hardware
//
#include <hip/hip_runtime.h>

// CDNA5 / gfx1250: bf16 WMMA mixed-precision sampled-logits GEMM.
// C[256 x 32768] = X[256 x 512] * gather(W, ids)^T + bias[ids]
// Pass 1: convert X to bf16 in workspace (done once).
// Pass 2: per block, gather+convert 64 sampled weight rows to LDS,
//         then 8 waves x (32 M-rows, 64 S-cols) of v_wmma_f32_16x16x32_bf16.

typedef __attribute__((ext_vector_type(16))) __bf16 v16bf;
typedef __attribute__((ext_vector_type(8)))  float  v8f;

#define D_DIM 512
#define N_ROW 256
#define S_DIM 32768
#define S_BLK 64          // sampled columns per block
// LDS row stride in bf16 elements: 536 -> 1072 B (16B aligned for ds_load_b128);
// 268 dwords, 268 mod 64 = 12, gcd = 4 => 16 row readers tile all 64 banks.
#define S_STR 536

// fp32 -> 2x bf16 (round-to-nearest-even), packed into one dword.
__device__ __forceinline__ unsigned int pack2bf(float lo, float hi) {
  union { float f; unsigned int u; } a, b;
  a.f = lo; b.f = hi;
  unsigned int ul = (a.u + 0x7FFFu + ((a.u >> 16) & 1u)) >> 16;
  unsigned int uh = (b.u + 0x7FFFu + ((b.u >> 16) & 1u)) & 0xFFFF0000u;
  return uh | ul;
}

union Frag16 { v16bf v; uint4 q[2]; };

// ---- Pass 1: X fp32 -> bf16 (256x512 = 131072 elements, 4 per thread) ----
__global__ __launch_bounds__(256) void cvt_x_bf16(const float* __restrict__ x,
                                                  unsigned short* __restrict__ xb) {
  const int i = (blockIdx.x * 256 + threadIdx.x) * 4;
  float4 f = *(const float4*)(x + i);
  uint2 p;
  p.x = pack2bf(f.x, f.y);
  p.y = pack2bf(f.z, f.w);
  *(uint2*)(xb + i) = p;
}

// ---- Pass 2: gathered GEMM ----
__global__ __launch_bounds__(256) void lsh_sampled_gemm(
    const unsigned short* __restrict__ xb,  // [256, 512] bf16
    const float* __restrict__ weight,       // [500001, 512] fp32
    const float* __restrict__ bias,         // [500001]
    const int*   __restrict__ ids,          // [32768]
    float*       __restrict__ out)          // [256, 32768]
{
  __shared__ __align__(16) unsigned short sW[S_BLK * S_STR]; // ~67 KB bf16 tile
  __shared__ float sBias[S_BLK];

  const int tid = threadIdx.x;
  const int s0  = blockIdx.x * S_BLK;

  // Gather 64 sampled weight rows, fp32 -> bf16, into LDS (once per S tile).
  {
    const int r = tid >> 2;          // row of the S tile (0..63)
    const int q = tid & 3;           // quarter of the 512-wide row
    const long long row = (long long)ids[s0 + r];
    const float* wr = weight + row * (long long)D_DIM + q * 128;
    unsigned short* dst = sW + r * S_STR + q * 128;
#pragma unroll
    for (int c = 0; c < 128; c += 4) {
      float4 f = *(const float4*)(wr + c);
      uint2 p;
      p.x = pack2bf(f.x, f.y);
      p.y = pack2bf(f.z, f.w);
      *(uint2*)(dst + c) = p;        // 8B LDS store, aligned (1072 % 8 == 0)
    }
    if (tid < S_BLK) sBias[tid] = bias[ids[s0 + tid]];
  }
  __syncthreads();

  // 8 waves; wave w owns M rows [32w, 32w+32) as two 16-row A fragments.
  const int wave  = tid >> 5;
  const int lane  = tid & 31;
  const int lhalf = lane >> 4;       // 0 or 1
  const int lmod  = lane & 15;

  // 16-bit A layout: lane%16 = M row; lanes 0-15 hold K {0..7,16..23},
  // lanes 16-31 hold K {8..15,24..31} of each 32-wide K slice.
  const unsigned short* xr0 = xb + (wave * 32 + lmod) * D_DIM;
  const unsigned short* xr1 = xr0 + 16 * D_DIM;
  const int koff = lhalf * 8;

  v8f acc00 = {}, acc01 = {}, acc02 = {}, acc03 = {};
  v8f acc10 = {}, acc11 = {}, acc12 = {}, acc13 = {};

  for (int kk = 0; kk < D_DIM; kk += 32) {
    Frag16 a0, a1;
    a0.q[0] = *(const uint4*)(xr0 + kk + koff);
    a0.q[1] = *(const uint4*)(xr0 + kk + koff + 16);
    a1.q[0] = *(const uint4*)(xr1 + kk + koff);
    a1.q[1] = *(const uint4*)(xr1 + kk + koff + 16);

    // B fragments from LDS: lane%16 = column, lanes 0-15 hold K {0..7,16..23}, lanes 16-31 K {8..15,24..31}.
    const int kb = kk + lhalf * 8;
    Frag16 b0, b1, b2, b3;
    const uint4* s;
    s = (const uint4*)(sW + (0 * 16 + lmod) * S_STR + kb); b0.q[0] = s[0]; b0.q[1] = s[2];
    s = (const uint4*)(sW + (1 * 16 + lmod) * S_STR + kb); b1.q[0] = s[0]; b1.q[1] = s[2];
    s = (const uint4*)(sW + (2 * 16 + lmod) * S_STR + kb); b2.q[0] = s[0]; b2.q[1] = s[2];
    s = (const uint4*)(sW + (3 * 16 + lmod) * S_STR + kb); b3.q[0] = s[0]; b3.q[1] = s[2];

    // (neg_a, A, neg_b, B, c_mod, C, reuse_a, reuse_b)
    acc00 = __builtin_amdgcn_wmma_f32_16x16x32_bf16(false, a0.v, false, b0.v, (short)0, acc00, false, false);
    acc10 = __builtin_amdgcn_wmma_f32_16x16x32_bf16(false, a1.v, false, b0.v, (short)0, acc10, false, false);
    acc01 = __builtin_amdgcn_wmma_f32_16x16x32_bf16(false, a0.v, false, b1.v, (short)0, acc01, false, false);
    acc11 = __builtin_amdgcn_wmma_f32_16x16x32_bf16(false, a1.v, false, b1.v, (short)0, acc11, false, false);
    acc02 = __builtin_amdgcn_wmma_f32_16x16x32_bf16(false, a0.v, false, b2.v, (short)0, acc02, false, false);
    acc12 = __builtin_amdgcn_wmma_f32_16x16x32_bf16(false, a1.v, false, b2.v, (short)0, acc12, false, false);
    acc03 = __builtin_amdgcn_wmma_f32_16x16x32_bf16(false, a0.v, false, b3.v, (short)0, acc03, false, false);
    acc13 = __builtin_amdgcn_wmma_f32_16x16x32_bf16(false, a1.v, false, b3.v, (short)0, acc13, false, false);
  }

  // Epilogue: C layout — VGPR i: lanes 0-15 row i, lanes 16-31 row 8+i.
  const float bb0 = sBias[ 0 + lmod];
  const float bb1 = sBias[16 + lmod];
  const float bb2 = sBias[32 + lmod];
  const float bb3 = sBias[48 + lmod];

  const int mrow0 = wave * 32 + lhalf * 8;          // A-fragment 0 rows
  float* op0 = out + (size_t)mrow0 * S_DIM + s0 + lmod;
  float* op1 = op0 + (size_t)16 * S_DIM;            // A-fragment 1 rows
#pragma unroll
  for (int i = 0; i < 8; ++i) {
    op0[(size_t)i * S_DIM +  0] = acc00[i] + bb0;
    op0[(size_t)i * S_DIM + 16] = acc01[i] + bb1;
    op0[(size_t)i * S_DIM + 32] = acc02[i] + bb2;
    op0[(size_t)i * S_DIM + 48] = acc03[i] + bb3;
    op1[(size_t)i * S_DIM +  0] = acc10[i] + bb0;
    op1[(size_t)i * S_DIM + 16] = acc11[i] + bb1;
    op1[(size_t)i * S_DIM + 32] = acc12[i] + bb2;
    op1[(size_t)i * S_DIM + 48] = acc13[i] + bb3;
  }
}

extern "C" void kernel_launch(void* const* d_in, const int* in_sizes, int n_in,
                              void* d_out, int out_size, void* d_ws, size_t ws_size,
                              hipStream_t stream) {
  (void)in_sizes; (void)n_in; (void)out_size; (void)ws_size;
  const float* x      = (const float*)d_in[0];
  const float* weight = (const float*)d_in[1];
  const float* bias   = (const float*)d_in[2];
  const int*   ids    = (const int*)d_in[3];
  float* out = (float*)d_out;
  unsigned short* xb = (unsigned short*)d_ws;   // 256x512 bf16 = 256 KB scratch

  // Pass 1: x fp32 -> bf16 (131072 elements, 4/thread, 128 blocks).
  cvt_x_bf16<<<dim3((N_ROW * D_DIM) / (256 * 4)), 256, 0, stream>>>(x, xb);
  // Pass 2: gathered GEMM; one block per 64-column S tile, full M=256.
  lsh_sampled_gemm<<<dim3(S_DIM / S_BLK), 256, 0, stream>>>(xb, weight, bias, ids, out);
}
